// DoubleStreamBlock_49890340110640
// MI455X (gfx1250) — matrix operand order repaired and checked by reading, not compiled, not run on hardware
//
#include <hip/hip_runtime.h>
#include <hip/hip_bf16.h>
#include <math.h>

typedef _Float16 half_t;
typedef __attribute__((ext_vector_type(4)))  _Float16 v4h;
typedef __attribute__((ext_vector_type(16))) _Float16 v16h;
typedef __attribute__((ext_vector_type(8)))  float    v8f;

#define TXT 256
#define IMG 2048
#define LTOT 2304
#define DMODEL 3072
#define NH 24
#define DH 128
#define MLPD 12288
#define MOD6 18432

// ---------------- workspace layout (bytes) ----------------
static constexpr size_t OFF_SV   = 0;                                    // 3072 f32
static constexpr size_t OFF_MOD  = 16384;                                // 2*18432 f32 (img then txt)
static constexpr size_t OFF_XN   = OFF_MOD + (size_t)2 * MOD6 * 4;       // LTOT*D f16 (ln1 then ln2)
static constexpr size_t OFF_QKV  = OFF_XN  + (size_t)LTOT * DMODEL * 2;  // LTOT*3D f32; reused as hidden f16
static constexpr size_t OFF_QH   = OFF_QKV + (size_t)LTOT * 3 * DMODEL * 4;
static constexpr size_t OFF_KH   = OFF_QH  + (size_t)NH * LTOT * DH * 2;
static constexpr size_t OFF_VH   = OFF_KH  + (size_t)NH * LTOT * DH * 2;
static constexpr size_t OFF_ATTN = OFF_VH  + (size_t)NH * LTOT * DH * 2; // LTOT*D f16

static __device__ __forceinline__ v8f wmma16(v16h a, v16h b, v8f c) {
  return __builtin_amdgcn_wmma_f32_16x16x32_f16(false, a, false, b, (short)0, c,
                                                false, false);
}

// CDNA5 async global->LDS copy (16B per lane), tracked by ASYNCcnt.
// LDS operand = low 32 bits of the generic shared pointer (flat->LDS per ISA 10.2).
static __device__ __forceinline__ void async_cp16(unsigned lds, const void* g) {
  asm volatile("global_load_async_to_lds_b128 %0, %1, off"
               :: "v"(lds), "v"((unsigned long long)(size_t)g)
               : "memory");
}
static __device__ __forceinline__ void wait_async0() {
  asm volatile("s_wait_asynccnt 0x0" ::: "memory");
}

// ---------------- tiny prologue kernels ----------------
__global__ void silu_kernel(const float* __restrict__ vec, float* __restrict__ sv) {
  int i = blockIdx.x * 256 + threadIdx.x;
  float v = vec[i];
  sv[i] = v / (1.0f + __expf(-v));
}

__global__ void mod_kernel(const float* __restrict__ sv,
                           const float* __restrict__ wi, const float* __restrict__ bi,
                           const float* __restrict__ wt, const float* __restrict__ bt,
                           float* __restrict__ mod) {
  int o = blockIdx.x * 256 + threadIdx.x;           // 0 .. 2*MOD6-1
  bool istxt = o >= MOD6;
  int oc = istxt ? (o - MOD6) : o;
  const float* w = istxt ? wt : wi;
  const float* b = istxt ? bt : bi;
  float acc = 0.f;
  for (int d = 0; d < DMODEL; ++d)
    acc = fmaf(sv[d], w[(size_t)d * MOD6 + oc], acc);
  mod[o] = acc + b[oc];
}

// LayerNorm + (1+scale)*x + shift  -> f16 token-major [LTOT][D]
__global__ __launch_bounds__(256) void ln_mod_kernel(
    const float* __restrict__ x_img, const float* __restrict__ x_txt,
    const float* __restrict__ mod, int sh_chunk, int sc_chunk,
    half_t* __restrict__ out) {
  __shared__ float red[8];
  const int l = blockIdx.x;
  const int tid = threadIdx.x;
  const int lane = tid & 31, wv = tid >> 5;
  const bool istxt = l < TXT;
  const float* x = istxt ? (x_txt + (size_t)l * DMODEL)
                         : (x_img + (size_t)(l - TXT) * DMODEL);
  const float* mv = mod + (istxt ? MOD6 : 0);
  const float* sh = mv + sh_chunk * DMODEL;
  const float* sc = mv + sc_chunk * DMODEL;

  float s = 0.f;
  for (int i = tid; i < DMODEL; i += 256) s += x[i];
  for (int o = 16; o > 0; o >>= 1) s += __shfl_xor(s, o);
  if (lane == 0) red[wv] = s;
  __syncthreads();
  float mu = 0.f;
  for (int j = 0; j < 8; ++j) mu += red[j];
  mu *= (1.0f / DMODEL);
  __syncthreads();
  float v = 0.f;
  for (int i = tid; i < DMODEL; i += 256) { float d = x[i] - mu; v += d * d; }
  for (int o = 16; o > 0; o >>= 1) v += __shfl_xor(v, o);
  if (lane == 0) red[wv] = v;
  __syncthreads();
  float var = 0.f;
  for (int j = 0; j < 8; ++j) var += red[j];
  var *= (1.0f / DMODEL);
  const float rstd = rsqrtf(var + 1e-6f);
  half_t* op = out + (size_t)l * DMODEL;
  for (int i = tid; i < DMODEL; i += 256)
    op[i] = (half_t)((x[i] - mu) * rstd * (1.0f + sc[i]) + sh[i]);
}

// ---------------- WMMA GEMM:  out = epilogue(A[f16 MxK] * W[f32 KxN] + bias) ----------------
// BM=128, BN=64, BK=32; 8 waves, each computing 16x64.
// A tile staged via async global->LDS; W tile staged transposed (Bt[n][k]) so
// B-fragment loads are two contiguous ds_load_b128 per WMMA.
enum { EP_PLAIN = 0, EP_GELU = 1, EP_GATERES = 2 };

template <int EPI>
__global__ __launch_bounds__(256, 2) void gemm_wmma_kernel(
    const half_t* __restrict__ A, const float* __restrict__ W,
    const float* __restrict__ bias, void* __restrict__ outp,
    const float* __restrict__ gate, const float* __restrict__ resid,
    int M, int N, int K) {
  __shared__ half_t As[128][40];
  __shared__ half_t Bt[64][40];  // transposed weight tile: [n][k]
  const int tid  = threadIdx.x;
  const int lane = tid & 31;
  const int wvi  = tid >> 5;                     // 0..7
  const int bm   = blockIdx.y * 128;
  const int bn   = blockIdx.x * 64;

  v8f acc[4];
  for (int f = 0; f < 4; ++f)
    for (int r = 0; r < 8; ++r) acc[f][r] = 0.0f;

  const int arow = tid >> 1, acol = (tid & 1) * 16;  // A tile: 128x32 f16 (async)
  const int nn   = (tid & 31) * 2, kk = (tid >> 5) * 4;  // W 4x2 sub-block
  const int fr   = wvi * 16 + (lane & 15);           // A-frag row
  const int koff = (lane >> 4) * 16;                 // A-frag K split
  const int n4   = lane & 15;                        // B/C column
  const int kb16 = (lane >> 4) * 16;                 // B-frag K split

  for (int k0 = 0; k0 < K; k0 += 32) {
    // A tile: async global->LDS (32B per thread)
    const half_t* ag = A + (size_t)(bm + arow) * K + k0 + acol;
    const unsigned al = (unsigned)(size_t)&As[arow][acol];
    async_cp16(al, ag);
    async_cp16(al + 16, (const char*)ag + 16);
    // W tile: load 4 rows x 2 cols f32, store transposed f16
    v4h c0, c1;
#pragma unroll
    for (int i = 0; i < 4; ++i) {
      const float2 w = *(const float2*)(W + (size_t)(k0 + kk + i) * N + bn + nn);
      c0[i] = (half_t)w.x;
      c1[i] = (half_t)w.y;
    }
    *(v4h*)&Bt[nn][kk]     = c0;
    *(v4h*)&Bt[nn + 1][kk] = c1;
    wait_async0();
    __syncthreads();

    v16h af;
#pragma unroll
    for (int i = 0; i < 8; ++i) {
      af[i]     = As[fr][koff + i];
      af[8 + i] = As[fr][koff + 8 + i];
    }
#pragma unroll
    for (int nf = 0; nf < 4; ++nf) {
      v16h bf;
      const half_t* bp = &Bt[nf * 16 + n4][kb16];
#pragma unroll
      for (int i = 0; i < 16; ++i) bf[i] = bp[i];
      acc[nf] = wmma16(af, bf, acc[nf]);
    }
    __syncthreads();
  }

  const int rbase = bm + wvi * 16 + (lane >> 4) * 8;
#pragma unroll
  for (int nf = 0; nf < 4; ++nf) {
    const int col = bn + nf * 16 + n4;
    const float bv = bias[col];
#pragma unroll
    for (int r = 0; r < 8; ++r) {
      const size_t idx = (size_t)(rbase + r) * N + col;
      float v = acc[nf][r] + bv;
      if (EPI == EP_PLAIN) {
        ((float*)outp)[idx] = v;
      } else if (EPI == EP_GELU) {
        float t = v + 0.044715f * v * v * v;
        float g = 0.5f * v * (1.0f + tanhf(0.7978845608028654f * t));
        ((half_t*)outp)[idx] = (half_t)g;
      } else {  // EP_GATERES
        ((float*)outp)[idx] = resid[idx] + v * gate[col];
      }
    }
  }
}

// ---------------- RMS norm + RoPE + head-major repack ----------------
__global__ __launch_bounds__(128) void rmsrope_kernel(
    const float* __restrict__ qkv, const float* __restrict__ pe,
    const float* __restrict__ iqs, const float* __restrict__ iks,
    const float* __restrict__ tqs, const float* __restrict__ tks,
    half_t* __restrict__ Qh, half_t* __restrict__ Kh, half_t* __restrict__ Vh) {
  const int l    = blockIdx.x;
  const int lane = threadIdx.x & 31;
  const int wv   = threadIdx.x >> 5;
  const int h    = blockIdx.y * 4 + wv;
  const bool istxt = (l < TXT);
  const float* qsc = istxt ? tqs : iqs;
  const float* ksc = istxt ? tks : iks;
  const size_t rowb = (size_t)l * (3 * DMODEL) + (size_t)h * DH;
  const float* peTok = pe + (size_t)l * 256;  // [64][2][2] per token
  const int d0 = lane * 4;
  const size_t ob = ((size_t)h * LTOT + l) * DH + d0;

  for (int s = 0; s < 2; ++s) {  // q then k
    const float* src = qkv + rowb + (size_t)s * DMODEL + d0;
    const float* scale = (s == 0) ? qsc : ksc;
    float x[4];
#pragma unroll
    for (int j = 0; j < 4; ++j) x[j] = src[j];
    float ss = x[0]*x[0] + x[1]*x[1] + x[2]*x[2] + x[3]*x[3];
    for (int o = 16; o > 0; o >>= 1) ss += __shfl_xor(ss, o);
    const float rms = rsqrtf(ss * (1.0f / DH) + 1e-6f);
#pragma unroll
    for (int j = 0; j < 4; ++j) x[j] *= rms * scale[d0 + j];
    float o4[4];
#pragma unroll
    for (int p = 0; p < 2; ++p) {
      const float* pb = peTok + (size_t)(2 * lane + p) * 4;
      float a = x[2 * p], b = x[2 * p + 1];
      o4[2 * p]     = pb[0] * a + pb[1] * b;
      o4[2 * p + 1] = pb[2] * a + pb[3] * b;
    }
    half_t* dst = ((s == 0) ? Qh : Kh) + ob;
#pragma unroll
    for (int j = 0; j < 4; ++j) dst[j] = (half_t)o4[j];
  }
  const float* vsrc = qkv + rowb + (size_t)2 * DMODEL + d0;
  half_t* vdst = Vh + ob;
#pragma unroll
  for (int j = 0; j < 4; ++j) vdst[j] = (half_t)vsrc[j];
}

// ---------------- attention (per head, 64-query block, 4 waves) ----------------
// K/V tiles double-buffered via async global->LDS copies (ASYNCcnt) so the DMA
// overlaps the 32 WMMA + exp of the current tile. No online max (RMS-normalized
// rows bound |score| <= sqrt(128)); per-lane partial row sums, one cross-lane
// reduction at the end.
__global__ __launch_bounds__(128, 1) void attn_kernel(
    const half_t* __restrict__ Qh, const half_t* __restrict__ Kh,
    const half_t* __restrict__ Vh, half_t* __restrict__ attn) {
  __shared__ half_t Ks[2][64][136];
  __shared__ half_t Vs[2][64][136];
  __shared__ half_t Ps[4][16][72];
  const int tid  = threadIdx.x;
  const int lane = tid & 31;
  const int wv   = tid >> 5;
  const int h    = blockIdx.y;
  const int q0   = blockIdx.x * 64 + wv * 16;
  const int koff = (lane >> 4) * 16;
  const int n4   = lane & 15;
  const int kb16 = (lane >> 4) * 16;
  const int ldr  = tid >> 1, ldc = (tid & 1) * 64;  // 64x128 coop tiles

  v16h qa[4];  // 16x128 query fragment
  {
    const half_t* qp = Qh + ((size_t)h * LTOT + q0 + n4) * DH;
#pragma unroll
    for (int c = 0; c < 4; ++c) {
      const half_t* p = qp + c * 32;
#pragma unroll
      for (int i = 0; i < 8; ++i) {
        qa[c][i]     = p[koff + i];
        qa[c][8 + i] = p[koff + 8 + i];
      }
    }
  }
  v8f O[8];
  for (int f = 0; f < 8; ++f)
    for (int r = 0; r < 8; ++r) O[f][r] = 0.f;
  float lsum[8];
  for (int r = 0; r < 8; ++r) lsum[r] = 0.f;
  const float scl = 0.08838834764831845f;  // 1/sqrt(128)

  auto issue = [&](int buf, int k0) {
    const size_t gb = ((size_t)h * LTOT + k0 + ldr) * DH + ldc;
    const char* kg = (const char*)(Kh + gb);
    const char* vg = (const char*)(Vh + gb);
    const unsigned kl = (unsigned)(size_t)&Ks[buf][ldr][ldc];
    const unsigned vl = (unsigned)(size_t)&Vs[buf][ldr][ldc];
#pragma unroll
    for (int j = 0; j < 8; ++j) {
      async_cp16(kl + j * 16, kg + j * 16);
      async_cp16(vl + j * 16, vg + j * 16);
    }
  };

  issue(0, 0);
  for (int k0 = 0, it = 0; k0 < LTOT; k0 += 64, ++it) {
    const int cb = it & 1;
    wait_async0();
    __syncthreads();
    if (k0 + 64 < LTOT) issue(cb ^ 1, k0 + 64);

    // S = Q * K^T for 16 rows x 64 keys
    v8f S[4];
#pragma unroll
    for (int ks = 0; ks < 4; ++ks) {
#pragma unroll
      for (int r = 0; r < 8; ++r) S[ks][r] = 0.f;
#pragma unroll
      for (int c = 0; c < 4; ++c) {
        v16h bf;
        const half_t* p = &Ks[cb][ks * 16 + n4][c * 32 + kb16];
#pragma unroll
        for (int i = 0; i < 16; ++i) bf[i] = p[i];
        S[ks] = wmma16(qa[c], bf, S[ks]);
      }
    }
    // P = exp(S*scale); per-lane partial row sums
    const int pr = (lane >> 4) * 8;
#pragma unroll
    for (int ks = 0; ks < 4; ++ks) {
#pragma unroll
      for (int r = 0; r < 8; ++r) {
        const float p = __expf(S[ks][r] * scl);
        lsum[r] += p;
        Ps[wv][pr + r][ks * 16 + n4] = (half_t)p;
      }
    }
    __syncthreads();
    // reload P in A-layout (two 16x32 fragments)
    v16h pa[2];
    {
      const half_t* p = &Ps[wv][n4][0];
#pragma unroll
      for (int u = 0; u < 2; ++u)
#pragma unroll
        for (int i = 0; i < 8; ++i) {
          pa[u][i]     = p[u * 32 + koff + i];
          pa[u][8 + i] = p[u * 32 + koff + 8 + i];
        }
    }
    // O += P * V
#pragma unroll
    for (int f = 0; f < 8; ++f) {
#pragma unroll
      for (int u = 0; u < 2; ++u) {
        v16h vb;
#pragma unroll
        for (int i = 0; i < 16; ++i)
          vb[i] = Vs[cb][u * 32 + kb16 + i][f * 16 + n4];
        O[f] = wmma16(pa[u], vb, O[f]);
      }
    }
    __syncthreads();
  }

  // single cross-lane reduction of row sums (16-lane groups)
#pragma unroll
  for (int r = 0; r < 8; ++r) {
    float s = lsum[r];
    s += __shfl_xor(s, 1);
    s += __shfl_xor(s, 2);
    s += __shfl_xor(s, 4);
    s += __shfl_xor(s, 8);
    lsum[r] = 1.0f / s;
  }
  const int rbase = (lane >> 4) * 8;
#pragma unroll
  for (int r = 0; r < 8; ++r) {
    const int tok = q0 + rbase + r;
    half_t* op = attn + (size_t)tok * DMODEL + h * DH;
#pragma unroll
    for (int f = 0; f < 8; ++f) op[f * 16 + n4] = (half_t)(O[f][r] * lsum[r]);
  }
}

// ---------------- host driver ----------------
extern "C" void kernel_launch(void* const* d_in, const int* in_sizes, int n_in,
                              void* d_out, int out_size, void* d_ws, size_t ws_size,
                              hipStream_t stream) {
  const float* img        = (const float*)d_in[0];
  const float* txt        = (const float*)d_in[1];
  const float* vec        = (const float*)d_in[2];
  const float* pe         = (const float*)d_in[3];
  const float* img_mod_w  = (const float*)d_in[4];
  const float* img_mod_b  = (const float*)d_in[5];
  const float* txt_mod_w  = (const float*)d_in[6];
  const float* txt_mod_b  = (const float*)d_in[7];
  const float* img_qkv_w  = (const float*)d_in[8];
  const float* img_qkv_b  = (const float*)d_in[9];
  const float* txt_qkv_w  = (const float*)d_in[10];
  const float* txt_qkv_b  = (const float*)d_in[11];
  const float* img_q_scale = (const float*)d_in[12];
  const float* img_k_scale = (const float*)d_in[13];
  const float* txt_q_scale = (const float*)d_in[14];
  const float* txt_k_scale = (const float*)d_in[15];
  const float* img_proj_w = (const float*)d_in[16];
  const float* img_proj_b = (const float*)d_in[17];
  const float* txt_proj_w = (const float*)d_in[18];
  const float* txt_proj_b = (const float*)d_in[19];
  const float* img_mlp_w1 = (const float*)d_in[20];
  const float* img_mlp_b1 = (const float*)d_in[21];
  const float* img_mlp_w2 = (const float*)d_in[22];
  const float* img_mlp_b2 = (const float*)d_in[23];
  const float* txt_mlp_w1 = (const float*)d_in[24];
  const float* txt_mlp_b1 = (const float*)d_in[25];
  const float* txt_mlp_w2 = (const float*)d_in[26];
  const float* txt_mlp_b2 = (const float*)d_in[27];

  char* ws = (char*)d_ws;
  float*  sv   = (float*)(ws + OFF_SV);
  float*  mod  = (float*)(ws + OFF_MOD);
  half_t* xn   = (half_t*)(ws + OFF_XN);
  float*  qkv  = (float*)(ws + OFF_QKV);
  half_t* hid  = (half_t*)(ws + OFF_QKV);  // reuse after qkv consumed
  half_t* Qh   = (half_t*)(ws + OFF_QH);
  half_t* Kh   = (half_t*)(ws + OFF_KH);
  half_t* Vh   = (half_t*)(ws + OFF_VH);
  half_t* attn = (half_t*)(ws + OFF_ATTN);
  float* out_img = (float*)d_out;                         // [IMG][D]
  float* out_txt = (float*)d_out + (size_t)IMG * DMODEL;  // [TXT][D]

  silu_kernel<<<DMODEL / 256, 256, 0, stream>>>(vec, sv);
  mod_kernel<<<(2 * MOD6) / 256, 256, 0, stream>>>(sv, img_mod_w, img_mod_b,
                                                   txt_mod_w, txt_mod_b, mod);
  // LN1 + mod (sh1=chunk0, sc1=chunk1)
  ln_mod_kernel<<<LTOT, 256, 0, stream>>>(img, txt, mod, 0, 1, xn);
  // QKV (txt rows 0..255, img rows 256..)
  gemm_wmma_kernel<EP_PLAIN><<<dim3((3 * DMODEL) / 64, TXT / 128), 256, 0, stream>>>(
      xn, txt_qkv_w, txt_qkv_b, qkv, nullptr, nullptr, TXT, 3 * DMODEL, DMODEL);
  gemm_wmma_kernel<EP_PLAIN><<<dim3((3 * DMODEL) / 64, IMG / 128), 256, 0, stream>>>(
      xn + (size_t)TXT * DMODEL, img_qkv_w, img_qkv_b,
      qkv + (size_t)TXT * 3 * DMODEL, nullptr, nullptr, IMG, 3 * DMODEL, DMODEL);
  // RMS + RoPE + repack
  rmsrope_kernel<<<dim3(LTOT, NH / 4), 128, 0, stream>>>(
      qkv, pe, img_q_scale, img_k_scale, txt_q_scale, txt_k_scale, Qh, Kh, Vh);
  // Attention
  attn_kernel<<<dim3(LTOT / 64, NH), 128, 0, stream>>>(Qh, Kh, Vh, attn);
  // proj + gate*x + residual  (g1 = chunk2)
  gemm_wmma_kernel<EP_GATERES><<<dim3(DMODEL / 64, TXT / 128), 256, 0, stream>>>(
      attn, txt_proj_w, txt_proj_b, out_txt, mod + MOD6 + 2 * DMODEL, txt,
      TXT, DMODEL, DMODEL);
  gemm_wmma_kernel<EP_GATERES><<<dim3(DMODEL / 64, IMG / 128), 256, 0, stream>>>(
      attn + (size_t)TXT * DMODEL, img_proj_w, img_proj_b, out_img,
      mod + 2 * DMODEL, img, IMG, DMODEL, DMODEL);
  // LN2 + mod (sh2=chunk3, sc2=chunk4); x1 lives in d_out
  ln_mod_kernel<<<LTOT, 256, 0, stream>>>(out_img, out_txt, mod, 3, 4, xn);
  // MLP up + GELU -> f16 hidden
  gemm_wmma_kernel<EP_GELU><<<dim3(MLPD / 64, TXT / 128), 256, 0, stream>>>(
      xn, txt_mlp_w1, txt_mlp_b1, hid, nullptr, nullptr, TXT, MLPD, DMODEL);
  gemm_wmma_kernel<EP_GELU><<<dim3(MLPD / 64, IMG / 128), 256, 0, stream>>>(
      xn + (size_t)TXT * DMODEL, img_mlp_w1, img_mlp_b1,
      hid + (size_t)TXT * MLPD, nullptr, nullptr, IMG, MLPD, DMODEL);
  // MLP down + gate + residual (g2 = chunk5), in-place on d_out
  gemm_wmma_kernel<EP_GATERES><<<dim3(DMODEL / 64, TXT / 128), 256, 0, stream>>>(
      hid, txt_mlp_w2, txt_mlp_b2, out_txt, mod + MOD6 + 5 * DMODEL, out_txt,
      TXT, DMODEL, MLPD);
  gemm_wmma_kernel<EP_GATERES><<<dim3(DMODEL / 64, IMG / 128), 256, 0, stream>>>(
      hid + (size_t)TXT * MLPD, img_mlp_w2, img_mlp_b2, out_img,
      mod + 5 * DMODEL, out_img, IMG, DMODEL, MLPD);
}
